// NeuralLongTermMemory_39041252720847
// MI455X (gfx1250) — compile-verified, matrix operand order checked by reading, not run on hardware
//
#include <hip/hip_runtime.h>

// ---------------------------------------------------------------------------
// Types
// ---------------------------------------------------------------------------
typedef __bf16 bf16_t;
typedef __bf16 v16b __attribute__((ext_vector_type(16)));
typedef float  v8f  __attribute__((ext_vector_type(8)));
typedef int    v4i_vs __attribute__((vector_size(16)));   // matches builtin param type

#define BM 128
#define BN 128
#define BK 32
#define LDSW 40   // u16 elements per LDS tile row (32 data + 8 pad) -> 80B rows, 16B aligned

enum {
  EPI_F32        = 0,  // outF = acc
  EPI_BF16       = 1,  // outB = bf16(acc)
  EPI_SILU_MIX   = 2,  // outF = acc (raw), outB = bf16(silu(acc))
  EPI_SILU_BF16  = 3,  // outB = bf16(silu(acc))
  EPI_DIFF_SCALE = 4,  // outB = bf16((acc - aux) * scale)
  EPI_DSILU      = 5,  // outB = bf16(acc * silu'(aux))
};

__device__ __forceinline__ float sigf(float x) { return 1.0f / (1.0f + __expf(-x)); }

union FragU { v16b v; float4 f4[2]; };   // 32B WMMA operand fragment (SROA-friendly)

// ---------------------------------------------------------------------------
// CDNA5 async memory->LDS copy (16B per lane) + asynccnt wait
// ---------------------------------------------------------------------------
#if __has_builtin(__builtin_amdgcn_global_load_async_to_lds_b128)
#define ASYNC_CP16(g, l)                                                      \
  __builtin_amdgcn_global_load_async_to_lds_b128(                             \
      (__attribute__((address_space(1))) v4i_vs*)(g),                         \
      (__attribute__((address_space(3))) v4i_vs*)(l), 0, 0)
#else
#define ASYNC_CP16(g, l)                                                      \
  asm volatile("global_load_async_to_lds_b128 %0, %1, off"                    \
               :: "v"((unsigned)(unsigned long long)                          \
                         (__attribute__((address_space(3))) void*)(l)),       \
                  "v"((unsigned long long)(const void*)(g))                   \
               : "memory")
#endif

#if __has_builtin(__builtin_amdgcn_s_wait_asynccnt)
#define WAIT_ASYNC(n) __builtin_amdgcn_s_wait_asynccnt(n)
#else
#define WAIT_ASYNC(n) asm volatile("s_wait_asynccnt " #n ::: "memory")
#endif

// ---------------------------------------------------------------------------
// NT GEMM: C[m,n] = sum_k A[m*lda+k] * B[n*ldb+k]    (X @ W^T form)
// Tiles stream to LDS via GLOBAL_LOAD_ASYNC_TO_LDS_B128, double-buffered.
// ---------------------------------------------------------------------------
template<int EPI>
__global__ __launch_bounds__(256)
void gemm_nt_wmma(const bf16_t* __restrict__ A, const bf16_t* __restrict__ B,
                  float* __restrict__ outF, bf16_t* __restrict__ outB,
                  const float* __restrict__ aux, float scale,
                  int M, int N, int K, int lda, int ldb, int ldc)
{
  __shared__ unsigned short As[2][BM * LDSW];
  __shared__ unsigned short Bs[2][BN * LDSW];

  const int tid  = threadIdx.x;
  const int lane = tid & 31;
  const int wave = tid >> 5;
  const int wm   = wave >> 2;           // 0..1  (64 rows of M per wave)
  const int wn   = wave & 3;            // 0..3  (32 cols of N per wave)
  const int bm   = blockIdx.y * BM;
  const int bn   = blockIdx.x * BN;
  const int fr   = lane & 15;
  const int hi   = lane >> 4;

  // staging geometry: thread -> 16 contiguous K-halves of one tile row
  const int ntRow  = tid >> 1;          // 0..127
  const int ntColh = (tid & 1) * 16;    // 0 or 16

  const char* gA0 = (const char*)(A + (size_t)(bm + ntRow) * lda + ntColh);
  const char* gB0 = (const char*)(B + (size_t)(bn + ntRow) * ldb + ntColh);
  unsigned short* lA[2] = { &As[0][ntRow * LDSW + ntColh], &As[1][ntRow * LDSW + ntColh] };
  unsigned short* lB[2] = { &Bs[0][ntRow * LDSW + ntColh], &Bs[1][ntRow * LDSW + ntColh] };

  auto issue = [&](int buf, int kt) {
    const char* ga = gA0 + (size_t)kt * (BK * 2);
    const char* gb = gB0 + (size_t)kt * (BK * 2);
    ASYNC_CP16(ga,      lA[buf]);
    ASYNC_CP16(ga + 16, lA[buf] + 8);
    ASYNC_CP16(gb,      lB[buf]);
    ASYNC_CP16(gb + 16, lB[buf] + 8);
  };

  v8f acc[4][2] = {};
  const int aRow0 = wm * 64;
  const int bRow0 = wn * 32;
  const int ktiles = K / BK;

  issue(0, 0);
  for (int kt = 0; kt < ktiles; ++kt) {
    const int buf = kt & 1;
    if (kt + 1 < ktiles) {
      issue(buf ^ 1, kt + 1);   // prefetch next tile into the other buffer
      WAIT_ASYNC(4);            // current buffer's 4 copies retired, prefetch in flight
    } else {
      WAIT_ASYNC(0);
    }
    __syncthreads();

    FragU af[4], bfr[2];
#pragma unroll
    for (int j = 0; j < 2; ++j) {
      const float4* p = (const float4*)&Bs[buf][(bRow0 + j * 16 + fr) * LDSW + hi * 8];
      bfr[j].f4[0] = p[0];
      bfr[j].f4[1] = p[2];
    }
#pragma unroll
    for (int i = 0; i < 4; ++i) {
      const float4* p = (const float4*)&As[buf][(aRow0 + i * 16 + fr) * LDSW + hi * 8];
      af[i].f4[0] = p[0];   // K = kbase..kbase+7
      af[i].f4[1] = p[2];   // K = 16+kbase..16+kbase+7
    }
#pragma unroll
    for (int i = 0; i < 4; ++i)
#pragma unroll
      for (int j = 0; j < 2; ++j)
        acc[i][j] = __builtin_amdgcn_wmma_f32_16x16x32_bf16(
            false, af[i].v, false, bfr[j].v, (short)0, acc[i][j], false, false);
    __syncthreads();
  }

  // Epilogue: C layout -> VGPR r holds M = r + 8*hi, lane fr = N
#pragma unroll
  for (int i = 0; i < 4; ++i) {
#pragma unroll
    for (int j = 0; j < 2; ++j) {
      const int mb = bm + wm * 64 + i * 16 + hi * 8;
      const int n  = bn + wn * 32 + j * 16 + fr;
#pragma unroll
      for (int r = 0; r < 8; ++r) {
        const size_t idx = (size_t)(mb + r) * ldc + n;
        const float v = acc[i][j][r];
        if (EPI == EPI_F32) {
          outF[idx] = v;
        } else if (EPI == EPI_BF16) {
          outB[idx] = (bf16_t)v;
        } else if (EPI == EPI_SILU_MIX) {
          outF[idx] = v;
          outB[idx] = (bf16_t)(v * sigf(v));
        } else if (EPI == EPI_SILU_BF16) {
          outB[idx] = (bf16_t)(v * sigf(v));
        } else if (EPI == EPI_DIFF_SCALE) {
          outB[idx] = (bf16_t)((v - aux[idx]) * scale);
        } else if (EPI == EPI_DSILU) {
          const float h = aux[idx];
          const float s = sigf(h);
          outB[idx] = (bf16_t)(v * (s * (1.0f + h * (1.0f - s))));
        }
      }
    }
  }
}

// ---------------------------------------------------------------------------
// TN GEMM: C[m,n] = sum_k A[k*lda+m] * B[k*ldb+n]   (A^T @ B, K = token dim)
// Tiles transposed on LDS store; staging registers are short-lived (no spill).
// ---------------------------------------------------------------------------
template<int EPI>
__global__ __launch_bounds__(256)
void gemm_tn_wmma(const bf16_t* __restrict__ A, const bf16_t* __restrict__ B,
                  float* __restrict__ outF, bf16_t* __restrict__ outB,
                  int M, int N, int K, int lda, int ldb, int ldc)
{
  __shared__ unsigned short As[BM * LDSW];
  __shared__ unsigned short Bs[BN * LDSW];

  const int tid  = threadIdx.x;
  const int lane = tid & 31;
  const int wave = tid >> 5;
  const int wm   = wave >> 2;
  const int wn   = wave & 3;
  const int bm   = blockIdx.y * BM;
  const int bn   = blockIdx.x * BN;
  const int fr   = lane & 15;
  const int hi   = lane >> 4;

  // staging: thread -> 16 contiguous M/N halves of one K-row
  const int tnKk = tid >> 3;            // 0..31
  const int tnMh = (tid & 7) * 16;      // 0..112

  v8f acc[4][2] = {};
  const int aRow0 = wm * 64;
  const int bRow0 = wn * 32;
  const int ktiles = K / BK;

  auto stash = [&](unsigned short* base, uint4 u0, uint4 u1) {
#define ST2(d, u)                                                         \
    do {                                                                  \
      base[(tnMh + 2 * (d)) * LDSW + tnKk]     = (unsigned short)((u) & 0xffffu); \
      base[(tnMh + 2 * (d) + 1) * LDSW + tnKk] = (unsigned short)((u) >> 16);     \
    } while (0)
    ST2(0, u0.x); ST2(1, u0.y); ST2(2, u0.z); ST2(3, u0.w);
    ST2(4, u1.x); ST2(5, u1.y); ST2(6, u1.z); ST2(7, u1.w);
#undef ST2
  };

  for (int kt = 0; kt < ktiles; ++kt) {
    const uint4* ga = (const uint4*)(A + (size_t)((size_t)kt * BK + tnKk) * lda + bm + tnMh);
    const uint4 a0 = ga[0], a1 = ga[1];
    const uint4* gb = (const uint4*)(B + (size_t)((size_t)kt * BK + tnKk) * ldb + bn + tnMh);
    const uint4 b0 = gb[0], b1 = gb[1];

    __syncthreads();                 // previous iteration's compute finished
    stash(As, a0, a1);
    stash(Bs, b0, b1);
    __syncthreads();

    FragU af[4], bfr[2];
#pragma unroll
    for (int j = 0; j < 2; ++j) {
      const float4* p = (const float4*)&Bs[(bRow0 + j * 16 + fr) * LDSW + hi * 8];
      bfr[j].f4[0] = p[0];
      bfr[j].f4[1] = p[2];
    }
#pragma unroll
    for (int i = 0; i < 4; ++i) {
      const float4* p = (const float4*)&As[(aRow0 + i * 16 + fr) * LDSW + hi * 8];
      af[i].f4[0] = p[0];
      af[i].f4[1] = p[2];
    }
#pragma unroll
    for (int i = 0; i < 4; ++i)
#pragma unroll
      for (int j = 0; j < 2; ++j)
        acc[i][j] = __builtin_amdgcn_wmma_f32_16x16x32_bf16(
            false, af[i].v, false, bfr[j].v, (short)0, acc[i][j], false, false);
  }

#pragma unroll
  for (int i = 0; i < 4; ++i) {
#pragma unroll
    for (int j = 0; j < 2; ++j) {
      const int mb = bm + wm * 64 + i * 16 + hi * 8;
      const int n  = bn + wn * 32 + j * 16 + fr;
#pragma unroll
      for (int r = 0; r < 8; ++r) {
        const size_t idx = (size_t)(mb + r) * ldc + n;
        const float v = acc[i][j][r];
        if (EPI == EPI_F32) outF[idx] = v;
        else                outB[idx] = (bf16_t)v;
      }
    }
  }
}

// ---------------------------------------------------------------------------
// Pointwise / reduction helpers (fp32 math, deterministic)
// ---------------------------------------------------------------------------
__global__ void k_cvt_bf16(const float* __restrict__ in, bf16_t* __restrict__ out, size_t n) {
  for (size_t i = (size_t)blockIdx.x * blockDim.x + threadIdx.x; i < n;
       i += (size_t)gridDim.x * blockDim.x)
    out[i] = (bf16_t)in[i];
}

__global__ void k_transpose_cvt(const float* __restrict__ in, bf16_t* __restrict__ out,
                                int rows, int cols) {
  const size_t n = (size_t)rows * cols;
  for (size_t i = (size_t)blockIdx.x * blockDim.x + threadIdx.x; i < n;
       i += (size_t)gridDim.x * blockDim.x) {
    const int r = (int)(i / cols);
    const int c = (int)(i % cols);
    out[(size_t)c * rows + r] = (bf16_t)in[i];
  }
}

__global__ void k_sig_partial(const float* __restrict__ z, const float* __restrict__ bias,
                              float* __restrict__ partial, size_t n, int dmask) {
  __shared__ float sm[256];
  float s = 0.0f;
  for (size_t i = (size_t)blockIdx.x * blockDim.x + threadIdx.x; i < n;
       i += (size_t)gridDim.x * blockDim.x)
    s += sigf(z[i] + bias[i & (size_t)dmask]);
  sm[threadIdx.x] = s;
  __syncthreads();
  for (int o = 128; o > 0; o >>= 1) {
    if ((int)threadIdx.x < o) sm[threadIdx.x] += sm[threadIdx.x + o];
    __syncthreads();
  }
  if (threadIdx.x == 0) partial[blockIdx.x] = sm[0];
}

__global__ void k_sig_final(const float* __restrict__ partial, int nb, float inv,
                            float* __restrict__ out) {
  if (threadIdx.x == 0 && blockIdx.x == 0) {
    float s = 0.0f;
    for (int i = 0; i < nb; ++i) s += partial[i];   // fixed order -> deterministic
    *out = s * inv;
  }
}

__global__ void k_update_w(const float* __restrict__ Wm, const float* __restrict__ S,
                           const float* __restrict__ g,
                           const float* __restrict__ alpha, const float* __restrict__ theta,
                           const float* __restrict__ eta,
                           bf16_t* __restrict__ out, size_t n) {
  const float a = 1.0f - *alpha, th = *theta, et = *eta;
  for (size_t i = (size_t)blockIdx.x * blockDim.x + threadIdx.x; i < n;
       i += (size_t)gridDim.x * blockDim.x)
    out[i] = (bf16_t)(a * Wm[i] + et * S[i] - th * g[i]);
}

// ---------------------------------------------------------------------------
// Host orchestration
// ---------------------------------------------------------------------------
extern "C" void kernel_launch(void* const* d_in, const int* in_sizes, int n_in,
                              void* d_out, int out_size, void* d_ws, size_t ws_size,
                              hipStream_t stream)
{
  (void)in_sizes; (void)n_in; (void)out_size; (void)ws_size;
  constexpr int NR = 2 * 8192;   // B*S tokens
  constexpr int D  = 1024;
  constexpr int H  = 2048;

  const float* x    = (const float*)d_in[0];
  const float* Wk   = (const float*)d_in[1];
  const float* Wv   = (const float*)d_in[2];
  const float* Wq   = (const float*)d_in[3];
  const float* Wout = (const float*)d_in[4];
  const float* Wgd  = (const float*)d_in[5];
  const float* bgd  = (const float*)d_in[6];
  const float* Wglr = (const float*)d_in[7];
  const float* bglr = (const float*)d_in[8];
  const float* Wgm  = (const float*)d_in[9];
  const float* bgm  = (const float*)d_in[10];
  const float* Wm1  = (const float*)d_in[11];
  const float* Wm2  = (const float*)d_in[12];
  const float* S1   = (const float*)d_in[13];
  const float* S2   = (const float*)d_in[14];
  float* out = (float*)d_out;

  char* wsb = (char*)d_ws;
  size_t off = 0;
  auto alloc = [&](size_t bytes) -> void* {
    void* p = wsb + off;
    off += (bytes + 255) & ~(size_t)255;
    return p;
  };

  bf16_t* xb    = (bf16_t*)alloc((size_t)NR * D * 2);
  bf16_t* kb    = (bf16_t*)alloc((size_t)NR * D * 2);
  bf16_t* qb    = (bf16_t*)alloc((size_t)NR * D * 2);
  float*  vf    = (float*) alloc((size_t)NR * D * 4);
  float*  zf    = (float*) alloc((size_t)NR * D * 4);   // gate pre-acts; later aliased by mem_out(bf16)
  float*  hf    = (float*) alloc((size_t)NR * H * 4);   // raw hidden; later aliased by a2 (bf16 silu)
  bf16_t* ab    = (bf16_t*)alloc((size_t)NR * H * 2);
  bf16_t* dpb   = (bf16_t*)alloc((size_t)NR * D * 2);
  bf16_t* dhb   = (bf16_t*)alloc((size_t)NR * H * 2);
  float*  g1f   = (float*) alloc((size_t)H * D * 4);
  float*  g2f   = (float*) alloc((size_t)D * H * 4);
  bf16_t* Wkb   = (bf16_t*)alloc((size_t)D * D * 2);
  bf16_t* Wvb   = (bf16_t*)alloc((size_t)D * D * 2);
  bf16_t* Wqb   = (bf16_t*)alloc((size_t)D * D * 2);
  bf16_t* Woutb = (bf16_t*)alloc((size_t)D * D * 2);
  bf16_t* Wgdb  = (bf16_t*)alloc((size_t)D * D * 2);
  bf16_t* Wglrb = (bf16_t*)alloc((size_t)D * D * 2);
  bf16_t* Wgmb  = (bf16_t*)alloc((size_t)D * D * 2);
  bf16_t* Wm1b  = (bf16_t*)alloc((size_t)H * D * 2);
  bf16_t* Wm2b  = (bf16_t*)alloc((size_t)D * H * 2);
  bf16_t* Wm2Tb = (bf16_t*)alloc((size_t)H * D * 2);
  bf16_t* W1nb  = (bf16_t*)alloc((size_t)H * D * 2);
  bf16_t* W2nb  = (bf16_t*)alloc((size_t)D * H * 2);
  float*  scal  = (float*) alloc(16);                   // alpha, theta, eta
  float*  part  = (float*) alloc(1024 * 4);

  bf16_t* a2b = (bf16_t*)hf;   // hf dead after dh epilogue
  bf16_t* mob = (bf16_t*)zf;   // zf dead after gate reductions

  const dim3 blk(256);
  const dim3 g1024(1024);
  const float invND = 1.0f / ((float)NR * (float)D);
  const float gscale = 2.0f * invND;   // d/dpred of mean squared error

  // ---- bf16 conversions of GEMM operands -------------------------------
  k_cvt_bf16<<<g1024, blk, 0, stream>>>(x,    xb,    (size_t)NR * D);
  k_cvt_bf16<<<g1024, blk, 0, stream>>>(Wk,   Wkb,   (size_t)D * D);
  k_cvt_bf16<<<g1024, blk, 0, stream>>>(Wv,   Wvb,   (size_t)D * D);
  k_cvt_bf16<<<g1024, blk, 0, stream>>>(Wq,   Wqb,   (size_t)D * D);
  k_cvt_bf16<<<g1024, blk, 0, stream>>>(Wout, Woutb, (size_t)D * D);
  k_cvt_bf16<<<g1024, blk, 0, stream>>>(Wgd,  Wgdb,  (size_t)D * D);
  k_cvt_bf16<<<g1024, blk, 0, stream>>>(Wglr, Wglrb, (size_t)D * D);
  k_cvt_bf16<<<g1024, blk, 0, stream>>>(Wgm,  Wgmb,  (size_t)D * D);
  k_cvt_bf16<<<g1024, blk, 0, stream>>>(Wm1,  Wm1b,  (size_t)H * D);
  k_cvt_bf16<<<g1024, blk, 0, stream>>>(Wm2,  Wm2b,  (size_t)D * H);
  k_transpose_cvt<<<g1024, blk, 0, stream>>>(Wm2, Wm2Tb, D, H);   // Wm2T: (H,D)

  // ---- projections: k, v, q --------------------------------------------
  gemm_nt_wmma<EPI_BF16><<<dim3(D / BN, NR / BM), blk, 0, stream>>>(
      xb, Wkb, nullptr, kb, nullptr, 0.f, NR, D, D, D, D, D);
  gemm_nt_wmma<EPI_F32><<<dim3(D / BN, NR / BM), blk, 0, stream>>>(
      xb, Wvb, vf, nullptr, nullptr, 0.f, NR, D, D, D, D, D);
  gemm_nt_wmma<EPI_BF16><<<dim3(D / BN, NR / BM), blk, 0, stream>>>(
      xb, Wqb, nullptr, qb, nullptr, 0.f, NR, D, D, D, D, D);

  // ---- gates: alpha, theta, eta = mean(sigmoid(x@Wg^T + b)) -------------
  gemm_nt_wmma<EPI_F32><<<dim3(D / BN, NR / BM), blk, 0, stream>>>(
      xb, Wgdb, zf, nullptr, nullptr, 0.f, NR, D, D, D, D, D);
  k_sig_partial<<<g1024, blk, 0, stream>>>(zf, bgd, part, (size_t)NR * D, D - 1);
  k_sig_final<<<1, 1, 0, stream>>>(part, 1024, invND, scal + 0);

  gemm_nt_wmma<EPI_F32><<<dim3(D / BN, NR / BM), blk, 0, stream>>>(
      xb, Wglrb, zf, nullptr, nullptr, 0.f, NR, D, D, D, D, D);
  k_sig_partial<<<g1024, blk, 0, stream>>>(zf, bglr, part, (size_t)NR * D, D - 1);
  k_sig_final<<<1, 1, 0, stream>>>(part, 1024, invND, scal + 1);

  gemm_nt_wmma<EPI_F32><<<dim3(D / BN, NR / BM), blk, 0, stream>>>(
      xb, Wgmb, zf, nullptr, nullptr, 0.f, NR, D, D, D, D, D);
  k_sig_partial<<<g1024, blk, 0, stream>>>(zf, bgm, part, (size_t)NR * D, D - 1);
  k_sig_final<<<1, 1, 0, stream>>>(part, 1024, invND, scal + 2);

  // ---- memory forward: h = k@Wm1^T ; a = silu(h) ------------------------
  gemm_nt_wmma<EPI_SILU_MIX><<<dim3(H / BN, NR / BM), blk, 0, stream>>>(
      kb, Wm1b, hf, ab, nullptr, 0.f, NR, H, D, D, D, H);
  // pred = a@Wm2^T ; dpred = 2*(pred - v)/count  (fused, pred never stored)
  gemm_nt_wmma<EPI_DIFF_SCALE><<<dim3(D / BN, NR / BM), blk, 0, stream>>>(
      ab, Wm2b, nullptr, dpb, vf, gscale, NR, D, H, H, H, D);

  // ---- backward: da = dpred@Wm2 ; dh = da * silu'(h)  (fused) -----------
  gemm_nt_wmma<EPI_DSILU><<<dim3(H / BN, NR / BM), blk, 0, stream>>>(
      dpb, Wm2Tb, nullptr, dhb, hf, 0.f, NR, H, D, D, D, H);
  // g1 = dh^T @ k : (H,D), contract over tokens
  gemm_tn_wmma<EPI_F32><<<dim3(D / BN, H / BM), blk, 0, stream>>>(
      dhb, kb, g1f, nullptr, H, D, NR, H, D, D);
  // g2 = dpred^T @ a : (D,H)
  gemm_tn_wmma<EPI_F32><<<dim3(H / BN, D / BM), blk, 0, stream>>>(
      dpb, ab, g2f, nullptr, D, H, NR, D, H, H);

  // ---- gated momentum + decay update (fp32 math, bf16 result) -----------
  k_update_w<<<g1024, blk, 0, stream>>>(Wm1, S1, g1f, scal + 0, scal + 1, scal + 2,
                                        W1nb, (size_t)H * D);
  k_update_w<<<g1024, blk, 0, stream>>>(Wm2, S2, g2f, scal + 0, scal + 1, scal + 2,
                                        W2nb, (size_t)D * H);

  // ---- retrieval + output projection ------------------------------------
  gemm_nt_wmma<EPI_SILU_BF16><<<dim3(H / BN, NR / BM), blk, 0, stream>>>(
      qb, W1nb, nullptr, a2b, nullptr, 0.f, NR, H, D, D, D, H);
  gemm_nt_wmma<EPI_BF16><<<dim3(D / BN, NR / BM), blk, 0, stream>>>(
      a2b, W2nb, nullptr, mob, nullptr, 0.f, NR, D, H, H, H, D);
  gemm_nt_wmma<EPI_F32><<<dim3(D / BN, NR / BM), blk, 0, stream>>>(
      mob, Woutb, out, nullptr, nullptr, 0.f, NR, D, D, D, D, D);
}
